// GraphAttentionLayer_4011499454635
// MI455X (gfx1250) — matrix operand order repaired and checked by reading, not compiled, not run on hardware
//
#include <hip/hip_runtime.h>

#define N_NODES 6144
#define IN_F    256
#define N_HEADS 4
#define N_HID   64
#define LOG2E   1.4426950408889634f
#define JHALF   (N_NODES / 2)

typedef __attribute__((ext_vector_type(16))) _Float16 v16h;
typedef __attribute__((ext_vector_type(8)))  float    v8f;

// p = exp2( leaky(ss+tt) - ss ) masked; everything pre-scaled by log2(e)
// (leaky-ReLU is positively homogeneous so the scaling commutes).
__device__ __forceinline__ _Float16 pval(int ad, float tg, float ss) {
  float e = ss + tg;
  e = fmaxf(e, 0.2f * e);
  const float p = exp2f(e - ss);
  return (_Float16)(ad > 0 ? p : 0.0f);
}

// ---------------------------------------------------------------------------
// Kernel 1: ht = h @ W (6144x256 @ 256x256) via v_wmma_f32_16x16x32_f16.
// Writes ht (f32 row-major) and htT (f16, [head][d][j]) for the P@V stage.
// ---------------------------------------------------------------------------
__global__ void __launch_bounds__(32)
gat_gemm_htW(const float* __restrict__ h, const float* __restrict__ W,
             float* __restrict__ ht, _Float16* __restrict__ htT) {
  const int lane = threadIdx.x & 31;
  const int hi = lane >> 4, lm = lane & 15;
  const int rowbase = blockIdx.x * 16;
  const int colbase = blockIdx.y * 16;
  const float* hrow = h + (rowbase + lm) * IN_F;

  v8f acc = {};
  for (int k0 = 0; k0 < IN_F; k0 += 32) {
    // A fragment: two contiguous 8-float runs per lane (K map 2v+8hi, +8 for v>=4)
    float av[16];
    *reinterpret_cast<float4*>(&av[0])  = *reinterpret_cast<const float4*>(&hrow[k0 + 8 * hi]);
    *reinterpret_cast<float4*>(&av[4])  = *reinterpret_cast<const float4*>(&hrow[k0 + 8 * hi + 4]);
    *reinterpret_cast<float4*>(&av[8])  = *reinterpret_cast<const float4*>(&hrow[k0 + 16 + 8 * hi]);
    *reinterpret_cast<float4*>(&av[12]) = *reinterpret_cast<const float4*>(&hrow[k0 + 16 + 8 * hi + 4]);
    v16h af, bf;
#pragma unroll
    for (int e = 0; e < 16; ++e) af[e] = (_Float16)av[e];
#pragma unroll
    for (int v = 0; v < 8; ++v) {  // B: K map 2v+8hi, +8 for v>=4, column gather from W
      const int kb = k0 + 2 * v + 8 * hi + (v >= 4 ? 8 : 0);
      bf[2 * v]     = (_Float16)W[kb * (N_HEADS * N_HID) + colbase + lm];
      bf[2 * v + 1] = (_Float16)W[(kb + 1) * (N_HEADS * N_HID) + colbase + lm];
    }
    acc = __builtin_amdgcn_wmma_f32_16x16x32_f16(false, af, false, bf,
                                                 (short)0, acc, false, false);
  }
#pragma unroll
  for (int v = 0; v < 8; ++v) {
    const int row = rowbase + v + 8 * hi;
    const int col = colbase + lm;
    const float val = acc[v];
    ht[row * (N_HEADS * N_HID) + col] = val;
    const int head = col >> 6, d = col & 63;
    htT[(head * N_HID + d) * N_NODES + row] = (_Float16)val;
  }
}

// ---------------------------------------------------------------------------
// Kernel 2: src/tgt[h][n] = log2(e) * dot(ht[n, h*64:(h+1)*64], a_src/tgt[h])
// ---------------------------------------------------------------------------
__global__ void gat_srctgt(const float* __restrict__ ht, const float* __restrict__ a,
                           float* __restrict__ src, float* __restrict__ tgt) {
  const int tid = blockIdx.x * blockDim.x + threadIdx.x;
  if (tid >= N_HEADS * N_NODES) return;
  const int head = tid / N_NODES, n = tid % N_NODES;
  const float* hp = ht + n * (N_HEADS * N_HID) + head * N_HID;
  const float* as = a + head * 2 * N_HID;
  float s = 0.f, t = 0.f;
#pragma unroll
  for (int d4 = 0; d4 < N_HID / 4; ++d4) {
    const float4 v = *reinterpret_cast<const float4*>(&hp[4 * d4]);
    s = fmaf(v.x, as[4 * d4 + 0], s);         t = fmaf(v.x, as[N_HID + 4 * d4 + 0], t);
    s = fmaf(v.y, as[4 * d4 + 1], s);         t = fmaf(v.y, as[N_HID + 4 * d4 + 1], t);
    s = fmaf(v.z, as[4 * d4 + 2], s);         t = fmaf(v.z, as[N_HID + 4 * d4 + 2], t);
    s = fmaf(v.w, as[4 * d4 + 3], s);         t = fmaf(v.w, as[N_HID + 4 * d4 + 3], t);
  }
  src[tid] = s * LOG2E;
  tgt[tid] = t * LOG2E;
}

// ---------------------------------------------------------------------------
// Kernel 3: fused masked softmax + P@ht. Block = 16 rows x half the j-range;
// 4 waves = 4 heads (adj tile shared via WGP$/L2, no LDS, no barriers).
// Unnormalized accumulators + row sums go to workspace; combined in kernel 4.
// Row sums come from a WMMA against an all-ones B, landing on the same
// (lane, vgpr) slots as the matching output rows.
// ---------------------------------------------------------------------------
__global__ void __launch_bounds__(128)
gat_attn(const int* __restrict__ adj, const float* __restrict__ src,
         const float* __restrict__ tgt, const _Float16* __restrict__ htT,
         float* __restrict__ pacc, float* __restrict__ pl) {
  const int head = threadIdx.x >> 5;
  const int lane = threadIdx.x & 31;
  const int hi = lane >> 4, lm = lane & 15;
  const int rowbase = blockIdx.x * 16;
  const int jbeg = blockIdx.y * JHALF;

  const float ss = src[head * N_NODES + rowbase + lm];
  const int* adjRow = adj + (size_t)(rowbase + lm) * N_NODES;
  const float* tgtH = tgt + head * N_NODES;
  const _Float16* htTH = htT + (size_t)head * N_HID * N_NODES;

  v8f acc[4] = {};
  v8f accL = {};
  v16h ones;
#pragma unroll
  for (int i = 0; i < 16; ++i) ones[i] = (_Float16)1.0f;

  for (int j0 = jbeg; j0 < jbeg + JHALF; j0 += 32) {
    __builtin_prefetch(&adjRow[j0 + 512], 0, 0);  // global_prefetch_b8: adj is the HBM stream
    const int off0 = j0 + 16 * hi;
    const int4   q0 = *reinterpret_cast<const int4*>(&adjRow[off0]);
    const int4   q1 = *reinterpret_cast<const int4*>(&adjRow[off0 + 4]);
    const int4   q2 = *reinterpret_cast<const int4*>(&adjRow[off0 + 8]);
    const int4   q3 = *reinterpret_cast<const int4*>(&adjRow[off0 + 12]);
    const float4 t0 = *reinterpret_cast<const float4*>(&tgtH[off0]);
    const float4 t1 = *reinterpret_cast<const float4*>(&tgtH[off0 + 4]);
    const float4 t2 = *reinterpret_cast<const float4*>(&tgtH[off0 + 8]);
    const float4 t3 = *reinterpret_cast<const float4*>(&tgtH[off0 + 12]);

    v16h af;
    af[0]  = pval(q0.x, t0.x, ss);  af[1]  = pval(q0.y, t0.y, ss);
    af[2]  = pval(q0.z, t0.z, ss);  af[3]  = pval(q0.w, t0.w, ss);
    af[4]  = pval(q1.x, t1.x, ss);  af[5]  = pval(q1.y, t1.y, ss);
    af[6]  = pval(q1.z, t1.z, ss);  af[7]  = pval(q1.w, t1.w, ss);
    af[8]  = pval(q2.x, t2.x, ss);  af[9]  = pval(q2.y, t2.y, ss);
    af[10] = pval(q2.z, t2.z, ss);  af[11] = pval(q2.w, t2.w, ss);
    af[12] = pval(q3.x, t3.x, ss);  af[13] = pval(q3.y, t3.y, ss);
    af[14] = pval(q3.z, t3.z, ss);  af[15] = pval(q3.w, t3.w, ss);

    accL = __builtin_amdgcn_wmma_f32_16x16x32_f16(false, af, false, ones,
                                                  (short)0, accL, false, false);
#pragma unroll
    for (int c = 0; c < 4; ++c) {
      // B fragment = 16 consecutive halfs of htT row (16c+lm): one v16h load
      const v16h bf = *reinterpret_cast<const v16h*>(
          &htTH[(size_t)(16 * c + lm) * N_NODES + j0 + 16 * hi]);
      acc[c] = __builtin_amdgcn_wmma_f32_16x16x32_f16(false, af, false, bf,
                                                      (short)0, acc[c], false, false);
    }
  }

  const int slot = blockIdx.y * N_HEADS + head;
#pragma unroll
  for (int v = 0; v < 8; ++v) {
    const int row = rowbase + v + 8 * hi;
#pragma unroll
    for (int c = 0; c < 4; ++c)
      pacc[((size_t)slot * N_NODES + row) * N_HID + 16 * c + lm] = acc[c][v];
  }
  if (lm == 0) {
#pragma unroll
    for (int v = 0; v < 8; ++v)
      pl[slot * N_NODES + rowbase + v + 8 * hi] = accL[v];
  }
}

// ---------------------------------------------------------------------------
// Kernel 4: combine j-halves, normalize, mean over heads
// ---------------------------------------------------------------------------
__global__ void gat_combine(const float* __restrict__ pacc, const float* __restrict__ pl,
                            float* __restrict__ out) {
  const int i = blockIdx.x * blockDim.x + threadIdx.x;
  if (i >= N_NODES * N_HID) return;
  const int n = i >> 6;
  float sum = 0.f;
#pragma unroll
  for (int hd = 0; hd < N_HEADS; ++hd) {
    const float l = pl[hd * N_NODES + n] + pl[(N_HEADS + hd) * N_NODES + n];
    const float v = pacc[(size_t)hd * N_NODES * N_HID + i] +
                    pacc[(size_t)(N_HEADS + hd) * N_NODES * N_HID + i];
    sum += v / fmaxf(l, 1e-30f);
  }
  out[i] = 0.25f * sum;
}

extern "C" void kernel_launch(void* const* d_in, const int* in_sizes, int n_in,
                              void* d_out, int out_size, void* d_ws, size_t ws_size,
                              hipStream_t stream) {
  (void)in_sizes; (void)n_in; (void)out_size; (void)ws_size;
  const float* h   = (const float*)d_in[0];
  const int*   adj = (const int*)d_in[1];
  const float* W   = (const float*)d_in[2];
  const float* a   = (const float*)d_in[3];
  float* out = (float*)d_out;

  char* ws = (char*)d_ws;
  float*    ht   = (float*)ws;                  // 6144*256 f32       = 6,291,456 B
  _Float16* htT  = (_Float16*)(ws + 6291456);   // 4*64*6144 f16      = 3,145,728 B
  float*    src  = (float*)(ws + 9437184);      // 4*6144 f32         =    98,304 B
  float*    tgt  = (float*)(ws + 9535488);      // 4*6144 f32         =    98,304 B
  float*    pl   = (float*)(ws + 9633792);      // 2*4*6144 f32       =   196,608 B
  float*    pacc = (float*)(ws + 9830400);      // 2*4*6144*64 f32    = 12,582,912 B

  gat_gemm_htW<<<dim3(N_NODES / 16, 16), 32, 0, stream>>>(h, W, ht, htT);
  gat_srctgt<<<(N_HEADS * N_NODES + 255) / 256, 256, 0, stream>>>(ht, a, src, tgt);
  gat_attn<<<dim3(N_NODES / 16, 2), 128, 0, stream>>>(adj, src, tgt, htT, pacc, pl);
  gat_combine<<<(N_NODES * N_HID + 255) / 256, 256, 0, stream>>>(pacc, pl, out);
}
